// Llama4Model_44710609551571
// MI455X (gfx1250) — matrix operand order repaired and checked by reading, not compiled, not run on hardware
//
#include <hip/hip_runtime.h>
#include <hip/hip_bf16.h>

// Problem constants (from reference): S=1024, B=2, D=1024, FF=2048, E=8
#define S_  1024
#define B_  2
#define T_  (S_ * B_)   // 2048 tokens
#define D_  1024
#define FF_ 2048
#define E_  8

typedef __bf16 bf16;
typedef __attribute__((ext_vector_type(16))) __bf16 v16bf;
typedef __attribute__((ext_vector_type(8)))  __bf16 v8bf;
typedef __attribute__((ext_vector_type(2)))  __bf16 v2bf;
typedef __attribute__((ext_vector_type(8)))  float  v8f;
typedef __attribute__((ext_vector_type(2)))  int    v2i;

// Workgroup tile: 256 threads = 8 wave32.  Wave grid 2(M) x 4(N).
#define BM  32
#define BN  128
#define KB  32
#define LDK 40   // padded K stride (bf16 elems) in LDS: 80B rows, 16B aligned

// ---------------------------------------------------------------------------
// CDNA5 async global->LDS copy (ASYNCcnt-tracked); verified present.
// ---------------------------------------------------------------------------
#if __has_builtin(__builtin_amdgcn_global_load_async_to_lds_b64)
#define HAVE_ASYNC_LDS 1
#endif

__device__ __forceinline__ void copy8_g2l(const bf16* g, bf16* l) {
#ifdef HAVE_ASYNC_LDS
  __builtin_amdgcn_global_load_async_to_lds_b64((v2i*)g, (v2i*)l, 0, 0);
#else
  *(uint2*)l = *(const uint2*)g;
#endif
}

__device__ __forceinline__ void wait_async_copies() {
#ifdef HAVE_ASYNC_LDS
#if __has_builtin(__builtin_amdgcn_s_wait_asynccnt)
  __builtin_amdgcn_s_wait_asynccnt(0);
#else
  asm volatile("s_wait_asynccnt 0" ::: "memory");
#endif
#endif
}

// Pack two fp32 into one bf16 pair and store as a single b32 to LDS
// (lowers to v_cvt_pk_bf16_f32 + ds_store_b32).
__device__ __forceinline__ void st_pair(bf16* dst, float a, float b) {
  v2bf p; p[0] = (bf16)a; p[1] = (bf16)b;
  *(v2bf*)dst = p;
}

// ---------------------------------------------------------------------------
// Fragment assembly helpers (contiguous K order, used for both A and B)
// A (16x32, MxK): lanes 0-15 row M=lane hold K 0..15;
//                 lanes 16-31 row M=lane-16 hold K 16..31.
// B (32x16, KxN): lane n=lane&15; lanes<16 hold K 0..15, lanes>=16 K 16..31.
// C/D (16x16 f32): elem r -> M = r + (lane>=16)*8, N = lane&15.
// ---------------------------------------------------------------------------
__device__ __forceinline__ v16bf ld_frag16(const bf16* p0, const bf16* p1) {
  v8bf lo = *(const v8bf*)p0;
  v8bf hi = *(const v8bf*)p1;
  v16bf r;
#pragma unroll
  for (int i = 0; i < 8; ++i) { r[i] = lo[i]; r[i + 8] = hi[i]; }
  return r;
}

__device__ __forceinline__ float silu_f(float g) {
  return g / (1.0f + __expf(-g));
}

// ---------------------------------------------------------------------------
// K0: zero per-expert counters
// ---------------------------------------------------------------------------
__global__ void k_zero_cnt(int* __restrict__ cnt) {
  if (threadIdx.x < E_) cnt[threadIdx.x] = 0;
}

// ---------------------------------------------------------------------------
// K1: router — one wave per token. logits = x . router_w[e], top-1 + sigmoid,
// and atomic bucketing into per-expert token lists.
// ---------------------------------------------------------------------------
__global__ void k_router(const float* __restrict__ x, const float* __restrict__ rw,
                         float* __restrict__ score, int* __restrict__ cnt,
                         int* __restrict__ list) {
  const int lane = threadIdx.x & 31;
  const int wv   = threadIdx.x >> 5;
  const int t    = blockIdx.x * 8 + wv;
  float acc[E_];
#pragma unroll
  for (int e = 0; e < E_; ++e) acc[e] = 0.0f;
  const float* xt = x + (size_t)t * D_;
  for (int j = lane; j < D_; j += 32) {
    float xv = xt[j];
#pragma unroll
    for (int e = 0; e < E_; ++e) acc[e] += xv * rw[e * D_ + j];
  }
#pragma unroll
  for (int e = 0; e < E_; ++e) {
#pragma unroll
    for (int off = 16; off > 0; off >>= 1) acc[e] += __shfl_xor(acc[e], off, 32);
  }
  if (lane == 0) {
    int best = 0; float bl = acc[0];
#pragma unroll
    for (int e = 1; e < E_; ++e) if (acc[e] > bl) { bl = acc[e]; best = e; }
    score[t] = 1.0f / (1.0f + __expf(-bl));
    int slot = atomicAdd(&cnt[best], 1);
    list[best * T_ + slot] = t;
  }
}

// ---------------------------------------------------------------------------
// K2: cast hidden states to bf16 (unscaled for shared expert, score-scaled
// for routed experts). 4 elements per thread.
// ---------------------------------------------------------------------------
__global__ void k_cast(const float* __restrict__ x, const float* __restrict__ score,
                       bf16* __restrict__ xb, bf16* __restrict__ xsb) {
  const int i    = blockIdx.x * 256 + threadIdx.x;
  const int base = i * 4;
  float4 v = *(const float4*)(x + base);
  float sc = score[base / D_];
  xb[base + 0] = (bf16)v.x;         xb[base + 1] = (bf16)v.y;
  xb[base + 2] = (bf16)v.z;         xb[base + 3] = (bf16)v.w;
  xsb[base + 0] = (bf16)(v.x * sc); xsb[base + 1] = (bf16)(v.y * sc);
  xsb[base + 2] = (bf16)(v.z * sc); xsb[base + 3] = (bf16)(v.w * sc);
}

// ---------------------------------------------------------------------------
// Weight-tile staging map: thread covers 2 adjacent K rows x 4 N cols per
// chunk (2 chunks).  idx = tid + c*256; k0 = (idx>>5)*2; n4 = (idx&31)*4.
// ---------------------------------------------------------------------------

// ---------------------------------------------------------------------------
// K3: shared expert up-projection, fused: H = silu(X@Wg) * (X@Wu)   (bf16 out)
// Pipelined: store tile k from regs -> barrier -> issue loads for tile k+1 ->
// WMMA on tile k -> barrier.  A tile is a pure bf16 copy -> async to LDS.
// ---------------------------------------------------------------------------
__global__ void k_mlp1_shared(const bf16* __restrict__ X, const float* __restrict__ Wg,
                              const float* __restrict__ Wu, bf16* __restrict__ H) {
  __shared__ bf16 sA [BM][LDK];
  __shared__ bf16 sBg[BN][LDK];
  __shared__ bf16 sBu[BN][LDK];
  const int tid  = threadIdx.x;
  const int lane = tid & 31;
  const int wv   = tid >> 5;
  const int wm   = wv & 1;
  const int wn   = wv >> 1;
  const int n0   = blockIdx.x * BN;
  const int m0   = blockIdx.y * BM;

  const int ra = tid >> 3, c4 = (tid & 7) << 2;      // A staging coords
  const bf16* Arow = X + (size_t)(m0 + ra) * D_ + c4;

  v8f cg[2], cu[2];
#pragma unroll
  for (int s = 0; s < 2; ++s)
#pragma unroll
    for (int i = 0; i < 8; ++i) { cg[s][i] = 0.0f; cu[s][i] = 0.0f; }

  // register-staged weight tiles (fp32): [chunk][krow]  (2 chunks x 2 rows)
  float4 rg[4], ru[4];
#pragma unroll
  for (int c = 0; c < 2; ++c) {
    int idx = tid + c * 256;
    int k0 = (idx >> 5) << 1, n4 = (idx & 31) << 2;
    rg[2*c+0] = *(const float4*)(Wg + (size_t)(k0 + 0) * FF_ + n0 + n4);
    rg[2*c+1] = *(const float4*)(Wg + (size_t)(k0 + 1) * FF_ + n0 + n4);
    ru[2*c+0] = *(const float4*)(Wu + (size_t)(k0 + 0) * FF_ + n0 + n4);
    ru[2*c+1] = *(const float4*)(Wu + (size_t)(k0 + 1) * FF_ + n0 + n4);
  }

  for (int kb = 0; kb < D_; kb += KB) {
    copy8_g2l(Arow + kb, &sA[ra][c4]);               // async A tile -> LDS
#pragma unroll
    for (int c = 0; c < 2; ++c) {
      int idx = tid + c * 256;
      int k0 = (idx >> 5) << 1, n4 = (idx & 31) << 2;
      st_pair(&sBg[n4 + 0][k0], rg[2*c].x, rg[2*c+1].x);
      st_pair(&sBg[n4 + 1][k0], rg[2*c].y, rg[2*c+1].y);
      st_pair(&sBg[n4 + 2][k0], rg[2*c].z, rg[2*c+1].z);
      st_pair(&sBg[n4 + 3][k0], rg[2*c].w, rg[2*c+1].w);
      st_pair(&sBu[n4 + 0][k0], ru[2*c].x, ru[2*c+1].x);
      st_pair(&sBu[n4 + 1][k0], ru[2*c].y, ru[2*c+1].y);
      st_pair(&sBu[n4 + 2][k0], ru[2*c].z, ru[2*c+1].z);
      st_pair(&sBu[n4 + 3][k0], ru[2*c].w, ru[2*c+1].w);
    }
    wait_async_copies();
    __syncthreads();

    if (kb + KB < D_) {                              // prefetch next tile
#pragma unroll
      for (int c = 0; c < 2; ++c) {
        int idx = tid + c * 256;
        int k0 = (idx >> 5) << 1, n4 = (idx & 31) << 2;
        rg[2*c+0] = *(const float4*)(Wg + (size_t)(kb + KB + k0 + 0) * FF_ + n0 + n4);
        rg[2*c+1] = *(const float4*)(Wg + (size_t)(kb + KB + k0 + 1) * FF_ + n0 + n4);
        ru[2*c+0] = *(const float4*)(Wu + (size_t)(kb + KB + k0 + 0) * FF_ + n0 + n4);
        ru[2*c+1] = *(const float4*)(Wu + (size_t)(kb + KB + k0 + 1) * FF_ + n0 + n4);
      }
    }

    const int ar = wm * 16 + (lane & 15);
    const int kl = (lane >> 4) * 8;
    const int kh = (lane >> 4) * 16;
    v16bf a = ld_frag16(&sA[ar][kh], &sA[ar][kh + 8]);
#pragma unroll
    for (int s = 0; s < 2; ++s) {
      int bn = wn * 32 + s * 16 + (lane & 15);
      v16bf bg = ld_frag16(&sBg[bn][kh], &sBg[bn][kh + 8]);
      v16bf bu = ld_frag16(&sBu[bn][kh], &sBu[bn][kh + 8]);
      cg[s] = __builtin_amdgcn_wmma_f32_16x16x32_bf16(false, a, false, bg, (short)0, cg[s], false, false);
      cu[s] = __builtin_amdgcn_wmma_f32_16x16x32_bf16(false, a, false, bu, (short)0, cu[s], false, false);
    }
    __syncthreads();
  }

#pragma unroll
  for (int s = 0; s < 2; ++s)
#pragma unroll
    for (int r = 0; r < 8; ++r) {
      float h = silu_f(cg[s][r]) * cu[s][r];
      int row = m0 + wm * 16 + (lane >> 4) * 8 + r;
      int col = n0 + wn * 32 + s * 16 + (lane & 15);
      H[(size_t)row * FF_ + col] = (bf16)h;
    }
}

// ---------------------------------------------------------------------------
// K4: shared expert down-projection: out = H @ Wd   (fp32 out, full overwrite)
// ---------------------------------------------------------------------------
__global__ void k_mlp2_shared(const bf16* __restrict__ H, const float* __restrict__ Wd,
                              float* __restrict__ out) {
  __shared__ bf16 sA[BM][LDK];
  __shared__ bf16 sB[BN][LDK];
  const int tid  = threadIdx.x;
  const int lane = tid & 31;
  const int wv   = tid >> 5;
  const int wm   = wv & 1;
  const int wn   = wv >> 1;
  const int n0   = blockIdx.x * BN;
  const int m0   = blockIdx.y * BM;

  const int ra = tid >> 3, c4 = (tid & 7) << 2;
  const bf16* Arow = H + (size_t)(m0 + ra) * FF_ + c4;

  v8f c[2];
#pragma unroll
  for (int s = 0; s < 2; ++s)
#pragma unroll
    for (int i = 0; i < 8; ++i) c[s][i] = 0.0f;

  float4 rd[4];
#pragma unroll
  for (int cc = 0; cc < 2; ++cc) {
    int idx = tid + cc * 256;
    int k0 = (idx >> 5) << 1, n4 = (idx & 31) << 2;
    rd[2*cc+0] = *(const float4*)(Wd + (size_t)(k0 + 0) * D_ + n0 + n4);
    rd[2*cc+1] = *(const float4*)(Wd + (size_t)(k0 + 1) * D_ + n0 + n4);
  }

  for (int kb = 0; kb < FF_; kb += KB) {
    copy8_g2l(Arow + kb, &sA[ra][c4]);
#pragma unroll
    for (int cc = 0; cc < 2; ++cc) {
      int idx = tid + cc * 256;
      int k0 = (idx >> 5) << 1, n4 = (idx & 31) << 2;
      st_pair(&sB[n4 + 0][k0], rd[2*cc].x, rd[2*cc+1].x);
      st_pair(&sB[n4 + 1][k0], rd[2*cc].y, rd[2*cc+1].y);
      st_pair(&sB[n4 + 2][k0], rd[2*cc].z, rd[2*cc+1].z);
      st_pair(&sB[n4 + 3][k0], rd[2*cc].w, rd[2*cc+1].w);
    }
    wait_async_copies();
    __syncthreads();

    if (kb + KB < FF_) {
#pragma unroll
      for (int cc = 0; cc < 2; ++cc) {
        int idx = tid + cc * 256;
        int k0 = (idx >> 5) << 1, n4 = (idx & 31) << 2;
        rd[2*cc+0] = *(const float4*)(Wd + (size_t)(kb + KB + k0 + 0) * D_ + n0 + n4);
        rd[2*cc+1] = *(const float4*)(Wd + (size_t)(kb + KB + k0 + 1) * D_ + n0 + n4);
      }
    }

    const int ar = wm * 16 + (lane & 15);
    const int kl = (lane >> 4) * 8;
    const int kh = (lane >> 4) * 16;
    v16bf a = ld_frag16(&sA[ar][kh], &sA[ar][kh + 8]);
#pragma unroll
    for (int s = 0; s < 2; ++s) {
      int bn = wn * 32 + s * 16 + (lane & 15);
      v16bf b = ld_frag16(&sB[bn][kh], &sB[bn][kh + 8]);
      c[s] = __builtin_amdgcn_wmma_f32_16x16x32_bf16(false, a, false, b, (short)0, c[s], false, false);
    }
    __syncthreads();
  }

#pragma unroll
  for (int s = 0; s < 2; ++s)
#pragma unroll
    for (int r = 0; r < 8; ++r) {
      int row = m0 + wm * 16 + (lane >> 4) * 8 + r;
      int col = n0 + wn * 32 + s * 16 + (lane & 15);
      out[(size_t)row * D_ + col] = c[s][r];
    }
}

// ---------------------------------------------------------------------------
// K5: routed expert up-projection (gathered rows via per-expert token list)
// ---------------------------------------------------------------------------
__global__ void k_mlp1_expert(const bf16* __restrict__ Xs, const float* __restrict__ WgA,
                              const float* __restrict__ WuA, bf16* __restrict__ H,
                              const int* __restrict__ list, const int* __restrict__ cnt) {
  const int e    = blockIdx.z;
  const int mcnt = cnt[e];
  const int m0   = blockIdx.y * BM;
  if (m0 >= mcnt) return;
  const float* Wg = WgA + (size_t)e * D_ * FF_;
  const float* Wu = WuA + (size_t)e * D_ * FF_;
  const int*   tl = list + e * T_;

  __shared__ bf16 sA [BM][LDK];
  __shared__ bf16 sBg[BN][LDK];
  __shared__ bf16 sBu[BN][LDK];
  const int tid  = threadIdx.x;
  const int lane = tid & 31;
  const int wv   = tid >> 5;
  const int wm   = wv & 1;
  const int wn   = wv >> 1;
  const int n0   = blockIdx.x * BN;

  const int ra = tid >> 3, c4 = (tid & 7) << 2;
  const bool arow_ok = (m0 + ra) < mcnt;
  const int  tok_a   = arow_ok ? tl[m0 + ra] : 0;
  const bf16* Arow   = Xs + (size_t)tok_a * D_ + c4;
  if (!arow_ok) { // zero-fill once; these LDS cells are never rewritten
    sA[ra][c4 + 0] = (bf16)0.0f; sA[ra][c4 + 1] = (bf16)0.0f;
    sA[ra][c4 + 2] = (bf16)0.0f; sA[ra][c4 + 3] = (bf16)0.0f;
  }

  v8f cg[2], cu[2];
#pragma unroll
  for (int s = 0; s < 2; ++s)
#pragma unroll
    for (int i = 0; i < 8; ++i) { cg[s][i] = 0.0f; cu[s][i] = 0.0f; }

  float4 rg[4], ru[4];
#pragma unroll
  for (int c = 0; c < 2; ++c) {
    int idx = tid + c * 256;
    int k0 = (idx >> 5) << 1, n4 = (idx & 31) << 2;
    rg[2*c+0] = *(const float4*)(Wg + (size_t)(k0 + 0) * FF_ + n0 + n4);
    rg[2*c+1] = *(const float4*)(Wg + (size_t)(k0 + 1) * FF_ + n0 + n4);
    ru[2*c+0] = *(const float4*)(Wu + (size_t)(k0 + 0) * FF_ + n0 + n4);
    ru[2*c+1] = *(const float4*)(Wu + (size_t)(k0 + 1) * FF_ + n0 + n4);
  }

  for (int kb = 0; kb < D_; kb += KB) {
    if (arow_ok) copy8_g2l(Arow + kb, &sA[ra][c4]);
#pragma unroll
    for (int c = 0; c < 2; ++c) {
      int idx = tid + c * 256;
      int k0 = (idx >> 5) << 1, n4 = (idx & 31) << 2;
      st_pair(&sBg[n4 + 0][k0], rg[2*c].x, rg[2*c+1].x);
      st_pair(&sBg[n4 + 1][k0], rg[2*c].y, rg[2*c+1].y);
      st_pair(&sBg[n4 + 2][k0], rg[2*c].z, rg[2*c+1].z);
      st_pair(&sBg[n4 + 3][k0], rg[2*c].w, rg[2*c+1].w);
      st_pair(&sBu[n4 + 0][k0], ru[2*c].x, ru[2*c+1].x);
      st_pair(&sBu[n4 + 1][k0], ru[2*c].y, ru[2*c+1].y);
      st_pair(&sBu[n4 + 2][k0], ru[2*c].z, ru[2*c+1].z);
      st_pair(&sBu[n4 + 3][k0], ru[2*c].w, ru[2*c+1].w);
    }
    wait_async_copies();
    __syncthreads();

    if (kb + KB < D_) {
#pragma unroll
      for (int c = 0; c < 2; ++c) {
        int idx = tid + c * 256;
        int k0 = (idx >> 5) << 1, n4 = (idx & 31) << 2;
        rg[2*c+0] = *(const float4*)(Wg + (size_t)(kb + KB + k0 + 0) * FF_ + n0 + n4);
        rg[2*c+1] = *(const float4*)(Wg + (size_t)(kb + KB + k0 + 1) * FF_ + n0 + n4);
        ru[2*c+0] = *(const float4*)(Wu + (size_t)(kb + KB + k0 + 0) * FF_ + n0 + n4);
        ru[2*c+1] = *(const float4*)(Wu + (size_t)(kb + KB + k0 + 1) * FF_ + n0 + n4);
      }
    }

    const int ar = wm * 16 + (lane & 15);
    const int kl = (lane >> 4) * 8;
    const int kh = (lane >> 4) * 16;
    v16bf a = ld_frag16(&sA[ar][kh], &sA[ar][kh + 8]);
#pragma unroll
    for (int s = 0; s < 2; ++s) {
      int bn = wn * 32 + s * 16 + (lane & 15);
      v16bf bg = ld_frag16(&sBg[bn][kh], &sBg[bn][kh + 8]);
      v16bf bu = ld_frag16(&sBu[bn][kh], &sBu[bn][kh + 8]);
      cg[s] = __builtin_amdgcn_wmma_f32_16x16x32_bf16(false, a, false, bg, (short)0, cg[s], false, false);
      cu[s] = __builtin_amdgcn_wmma_f32_16x16x32_bf16(false, a, false, bu, (short)0, cu[s], false, false);
    }
    __syncthreads();
  }

#pragma unroll
  for (int s = 0; s < 2; ++s)
#pragma unroll
    for (int r = 0; r < 8; ++r) {
      int slot = m0 + wm * 16 + (lane >> 4) * 8 + r;
      if (slot < mcnt) {
        int tok = tl[slot];
        float h = silu_f(cg[s][r]) * cu[s][r];
        int col = n0 + wn * 32 + s * 16 + (lane & 15);
        H[(size_t)tok * FF_ + col] = (bf16)h;
      }
    }
}

// ---------------------------------------------------------------------------
// K6: routed expert down-projection; accumulates into out (rows unique/token,
// and this kernel runs after K4 on the same stream, so plain RMW is safe).
// ---------------------------------------------------------------------------
__global__ void k_mlp2_expert(const bf16* __restrict__ H, const float* __restrict__ WdA,
                              float* __restrict__ out, const int* __restrict__ list,
                              const int* __restrict__ cnt) {
  const int e    = blockIdx.z;
  const int mcnt = cnt[e];
  const int m0   = blockIdx.y * BM;
  if (m0 >= mcnt) return;
  const float* Wd = WdA + (size_t)e * FF_ * D_;
  const int*   tl = list + e * T_;

  __shared__ bf16 sA[BM][LDK];
  __shared__ bf16 sB[BN][LDK];
  const int tid  = threadIdx.x;
  const int lane = tid & 31;
  const int wv   = tid >> 5;
  const int wm   = wv & 1;
  const int wn   = wv >> 1;
  const int n0   = blockIdx.x * BN;

  const int ra = tid >> 3, c4 = (tid & 7) << 2;
  const bool arow_ok = (m0 + ra) < mcnt;
  const int  tok_a   = arow_ok ? tl[m0 + ra] : 0;
  const bf16* Arow   = H + (size_t)tok_a * FF_ + c4;
  if (!arow_ok) {
    sA[ra][c4 + 0] = (bf16)0.0f; sA[ra][c4 + 1] = (bf16)0.0f;
    sA[ra][c4 + 2] = (bf16)0.0f; sA[ra][c4 + 3] = (bf16)0.0f;
  }

  v8f c[2];
#pragma unroll
  for (int s = 0; s < 2; ++s)
#pragma unroll
    for (int i = 0; i < 8; ++i) c[s][i] = 0.0f;

  float4 rd[4];
#pragma unroll
  for (int cc = 0; cc < 2; ++cc) {
    int idx = tid + cc * 256;
    int k0 = (idx >> 5) << 1, n4 = (idx & 31) << 2;
    rd[2*cc+0] = *(const float4*)(Wd + (size_t)(k0 + 0) * D_ + n0 + n4);
    rd[2*cc+1] = *(const float4*)(Wd + (size_t)(k0 + 1) * D_ + n0 + n4);
  }

  for (int kb = 0; kb < FF_; kb += KB) {
    if (arow_ok) copy8_g2l(Arow + kb, &sA[ra][c4]);
#pragma unroll
    for (int cc = 0; cc < 2; ++cc) {
      int idx = tid + cc * 256;
      int k0 = (idx >> 5) << 1, n4 = (idx & 31) << 2;
      st_pair(&sB[n4 + 0][k0], rd[2*cc].x, rd[2*cc+1].x);
      st_pair(&sB[n4 + 1][k0], rd[2*cc].y, rd[2*cc+1].y);
      st_pair(&sB[n4 + 2][k0], rd[2*cc].z, rd[2*cc+1].z);
      st_pair(&sB[n4 + 3][k0], rd[2*cc].w, rd[2*cc+1].w);
    }
    wait_async_copies();
    __syncthreads();

    if (kb + KB < FF_) {
#pragma unroll
      for (int cc = 0; cc < 2; ++cc) {
        int idx = tid + cc * 256;
        int k0 = (idx >> 5) << 1, n4 = (idx & 31) << 2;
        rd[2*cc+0] = *(const float4*)(Wd + (size_t)(kb + KB + k0 + 0) * D_ + n0 + n4);
        rd[2*cc+1] = *(const float4*)(Wd + (size_t)(kb + KB + k0 + 1) * D_ + n0 + n4);
      }
    }

    const int ar = wm * 16 + (lane & 15);
    const int kl = (lane >> 4) * 8;
    const int kh = (lane >> 4) * 16;
    v16bf a = ld_frag16(&sA[ar][kh], &sA[ar][kh + 8]);
#pragma unroll
    for (int s = 0; s < 2; ++s) {
      int bn = wn * 32 + s * 16 + (lane & 15);
      v16bf b = ld_frag16(&sB[bn][kh], &sB[bn][kh + 8]);
      c[s] = __builtin_amdgcn_wmma_f32_16x16x32_bf16(false, a, false, b, (short)0, c[s], false, false);
    }
    __syncthreads();
  }

#pragma unroll
  for (int s = 0; s < 2; ++s)
#pragma unroll
    for (int r = 0; r < 8; ++r) {
      int slot = m0 + wm * 16 + (lane >> 4) * 8 + r;
      if (slot < mcnt) {
        int tok = tl[slot];
        int col = n0 + wn * 32 + s * 16 + (lane & 15);
        size_t o = (size_t)tok * D_ + col;
        out[o] = out[o] + c[s][r];
      }
    }
}

// ---------------------------------------------------------------------------
extern "C" void kernel_launch(void* const* d_in, const int* in_sizes, int n_in,
                              void* d_out, int out_size, void* d_ws, size_t ws_size,
                              hipStream_t stream) {
  (void)in_sizes; (void)n_in; (void)out_size; (void)ws_size;
  const float* x  = (const float*)d_in[0];  // [S,B,D]
  const float* rw = (const float*)d_in[1];  // [E,D]
  const float* eg = (const float*)d_in[2];  // [E,D,FF]
  const float* eu = (const float*)d_in[3];  // [E,D,FF]
  const float* ed = (const float*)d_in[4];  // [E,FF,D]
  const float* sg = (const float*)d_in[5];  // [D,FF]
  const float* su = (const float*)d_in[6];  // [D,FF]
  const float* sd = (const float*)d_in[7];  // [FF,D]
  float* out = (float*)d_out;               // [S,B,D]

  char* ws = (char*)d_ws;
  size_t off = 0;
  bf16*  xb    = (bf16*)(ws + off); off += (size_t)T_ * D_ * 2;   // 4 MB
  bf16*  xsb   = (bf16*)(ws + off); off += (size_t)T_ * D_ * 2;   // 4 MB
  bf16*  Hs    = (bf16*)(ws + off); off += (size_t)T_ * FF_ * 2;  // 8 MB
  bf16*  He    = (bf16*)(ws + off); off += (size_t)T_ * FF_ * 2;  // 8 MB
  float* score = (float*)(ws + off); off += (size_t)T_ * 4;
  int*   cnt   = (int*)(ws + off);   off += 256;
  int*   list  = (int*)(ws + off);   off += (size_t)E_ * T_ * 4;

  k_zero_cnt<<<1, 32, 0, stream>>>(cnt);
  k_router<<<T_ / 8, 256, 0, stream>>>(x, rw, score, cnt, list);
  k_cast<<<(T_ * D_ / 4) / 256, 256, 0, stream>>>(x, score, xb, xsb);
  k_mlp1_shared<<<dim3(FF_ / BN, T_ / BM), 256, 0, stream>>>(xb, sg, su, Hs);
  k_mlp2_shared<<<dim3(D_ / BN, T_ / BM), 256, 0, stream>>>(Hs, sd, out);
  k_mlp1_expert<<<dim3(FF_ / BN, T_ / BM, E_), 256, 0, stream>>>(xsb, eg, eu, He, list, cnt);
  k_mlp2_expert<<<dim3(D_ / BN, T_ / BM, E_), 256, 0, stream>>>(He, ed, out, list, cnt);
}
